// ShiftedNeighborhoodCompensationBlock_31404800868743
// MI455X (gfx1250) — matrix operand order repaired and checked by reading, not compiled, not run on hardware
//
#include <hip/hip_runtime.h>
#include <hip/hip_bf16.h>

// ---------------------------------------------------------------------------
// Problem constants (B=2, C=64, H=W=96)
// ---------------------------------------------------------------------------
#define BB   2
#define CC   64
#define HH   96
#define WW   96
#define NPIX (BB * HH * WW)      // 18432 pixels
#define KRED 128                 // reduce GEMM K
#define KFUSE 1728               // fuse GEMM K = 9 taps * 192 channels

typedef __attribute__((ext_vector_type(16))) __bf16 v16bf;
typedef __attribute__((ext_vector_type(8)))  float  v8f;

static __device__ __forceinline__ unsigned short f2bf(float f) {
  union { float f; unsigned u; } x; x.f = f;
  unsigned r = x.u + 0x7FFFu + ((x.u >> 16) & 1u);   // round-to-nearest-even
  return (unsigned short)(r >> 16);
}
static __device__ __forceinline__ int imin(int a, int b) { return a < b ? a : b; }
static __device__ __forceinline__ int imax(int a, int b) { return a > b ? a : b; }

// ---------------------------------------------------------------------------
// Kernel 1: shifted grouped 3x3 conv ("smooth") + pack [past || smooth] as
// bf16 NHWC-concat [N][128]. Channels 0..63 = past (reused by fuse GEMM).
// shifted[c][y][x] = past[c][y - sy][x - sx] with zero pad on either stage.
// ---------------------------------------------------------------------------
__global__ void k_shift_smooth_pack(const float* __restrict__ past,
                                    const float* __restrict__ w_smooth,
                                    const float* __restrict__ b_smooth,
                                    unsigned short* __restrict__ Xcat) {
  int idx = blockIdx.x * blockDim.x + threadIdx.x;
  if (idx >= BB * CC * HH * WW) return;
  int x = idx % WW; int t = idx / WW;
  int y = t % HH;   t /= HH;
  int o = t % CC;   int b = t / CC;
  int g = o >> 3;
  const int sy[8] = {-1,-1,-1, 0, 0, 1, 1, 1};
  const int sx[8] = {-1, 0, 1,-1, 1,-1, 0, 1};
  float acc = b_smooth[o];
  for (int i = 0; i < 8; ++i) {
    const float* pc = past + (((long)b * CC + (g * 8 + i)) * HH) * WW;
    for (int ky = 0; ky < 3; ++ky) {
      int yy = y + ky - 1;                 // conv pad of shifted
      if ((unsigned)yy >= (unsigned)HH) continue;
      int ys = yy - sy[g];                 // roll-of-padded => shift w/ zero pad
      if ((unsigned)ys >= (unsigned)HH) continue;
      for (int kx = 0; kx < 3; ++kx) {
        int xx = x + kx - 1;
        if ((unsigned)xx >= (unsigned)WW) continue;
        int xs = xx - sx[g];
        if ((unsigned)xs >= (unsigned)WW) continue;
        acc += w_smooth[((o * 8 + i) * 3 + ky) * 3 + kx] * pc[ys * WW + xs];
      }
    }
  }
  long n = ((long)b * HH + y) * WW + x;
  Xcat[n * KRED + 64 + o] = f2bf(acc);
  Xcat[n * KRED + o]      = f2bf(past[(((long)b * CC + o) * HH + y) * WW + x]);
}

// ---------------------------------------------------------------------------
// Weight packers
// ---------------------------------------------------------------------------
__global__ void k_f32_to_bf16(const float* __restrict__ src,
                              unsigned short* __restrict__ dst, int n) {
  int i = blockIdx.x * blockDim.x + threadIdx.x;
  if (i < n) dst[i] = f2bf(src[i]);
}

// w_fuse [M=64][Cin=192][3][3]  ->  Wf[m][k],  k = (ky*3+kx)*192 + c
__global__ void k_pack_wfuse(const float* __restrict__ w,
                             unsigned short* __restrict__ Wf) {
  int idx = blockIdx.x * blockDim.x + threadIdx.x;
  if (idx >= 64 * KFUSE) return;
  int m = idx / KFUSE, k = idx % KFUSE;
  int tap = k / 192, c = k % 192;
  int ky = tap / 3, kx = tap % 3;
  Wf[idx] = f2bf(w[(((long)m * 192 + c) * 3 + ky) * 3 + kx]);
}

// ---------------------------------------------------------------------------
// Wide bf16 GEMM: each wave computes a 16-pixel x 64-channel strip.
//   out[i][j] = bias[j] + sum_k A[i][k] * Wt[j][k]     (Cout fixed at 64)
// Per K-step: one A operand load, four B tiles, four independent WMMAs
// (4 accumulators -> amortizes A traffic 4x and hides WMMA latency).
// blockDim = 128 (4 waves, each an independent strip).
// A/B k order: contiguous k = 16*half + j in both operands.
// ---------------------------------------------------------------------------
__global__ void k_gemm_bf16_wide(const unsigned short* __restrict__ A, int sA,
                                 const unsigned short* __restrict__ Wt,
                                 const float* __restrict__ bias,
                                 float* __restrict__ outF,
                                 unsigned short* __restrict__ outB,
                                 int K) {
  int wave  = threadIdx.x >> 5;
  int ti    = (blockIdx.x * 4 + wave) << 4;      // 16-pixel strip base
  int lane  = threadIdx.x & 31;
  int half  = lane >> 4, lan16 = lane & 15;
  v8f acc[4] = {};
  for (int k0 = 0; k0 < K; k0 += 32) {
    union { unsigned short u[16]; v16bf v; } am, bm[4];
    const unsigned short* arow = A + (long)(ti + lan16) * sA + k0;
#pragma unroll
    for (int j = 0; j < 16; ++j) {
      int tk = half * 16 + j;   // A K-strip contiguous, matches B
      am.u[j] = arow[tk];
    }
    const unsigned short* bbase = Wt + (long)lan16 * K + k0 + half * 16;
#pragma unroll
    for (int t = 0; t < 4; ++t) {
#pragma unroll
      for (int j = 0; j < 16; ++j) bm[t].u[j] = bbase[(long)t * 16 * K + j];
    }
#pragma unroll
    for (int t = 0; t < 4; ++t)
      acc[t] = __builtin_amdgcn_wmma_f32_16x16x32_bf16(false, am.v, false, bm[t].v,
                                                       (short)0, acc[t], false, false);
  }
#pragma unroll
  for (int t = 0; t < 4; ++t) {
    int j = t * 16 + lan16;
    float bv = bias ? bias[j] : 0.f;
#pragma unroll
    for (int r = 0; r < 8; ++r) {
      long i = ti + r + 8 * half;                    // D: m = r + 8*(lane>=16)
      float v = acc[t][r] + bv;
      if (outF) outF[i * CC + j] = v;
      if (outB) outB[i * CC + j] = f2bf(v);
    }
  }
}

// ---------------------------------------------------------------------------
// Neighborhood attention, one 64-thread block per pixel (thread = channel).
// Q from curr (NCHW f32), K = past2 NHWC f32, V NHWC f32. Output bf16 NHWC.
// ---------------------------------------------------------------------------
template <int KS>
__global__ void k_natten(const float* __restrict__ curr,
                         const float* __restrict__ Kf,
                         const float* __restrict__ Vf,
                         const float* __restrict__ rpb,
                         unsigned short* __restrict__ outB) {
  constexpr int K2 = KS * KS;
  constexpr int R  = KS / 2;
  long n = blockIdx.x;
  int c = threadIdx.x;
  int w = (int)(n % WW); long t = n / WW;
  int h = (int)(t % HH); int b = (int)(t / HH);
  __shared__ float qs[CC];
  __shared__ float sc[K2];
  qs[c] = curr[(((long)b * CC + c) * HH + h) * WW + w];
  __syncthreads();
  int h0 = imin(imax(h - R, 0), HH - KS);
  int w0 = imin(imax(w - R, 0), WW - KS);
  if (c < K2) {
    int i = c / KS, j = c % KS;
    long np = ((long)b * HH + (h0 + i)) * WW + (w0 + j);
    const float* krow = Kf + np * CC;
    float s = 0.f;
    for (int cc = 0; cc < CC; ++cc) s += qs[cc] * krow[cc];
    s += rpb[(h0 + i - h + KS - 1) * (2 * KS - 1) + (w0 + j - w + KS - 1)];
    sc[c] = s;
  }
  __syncthreads();
  float mx = -3.4e38f;
  for (int q = 0; q < K2; ++q) mx = fmaxf(mx, sc[q]);
  float sum = 0.f;
  for (int q = 0; q < K2; ++q) sum += __expf(sc[q] - mx);
  float inv = 1.f / sum, acc = 0.f;
  for (int q = 0; q < K2; ++q) {
    int i = q / KS, j = q % KS;
    long np = ((long)b * HH + (h0 + i)) * WW + (w0 + j);
    acc += __expf(sc[q] - mx) * inv * Vf[np * CC + c];
  }
  outB[n * CC + c] = f2bf(acc);
}

// ---------------------------------------------------------------------------
// Fuse 3x3 conv as implicit im2col GEMM, wide form: each wave computes a
// 16-pixel x 64-channel strip over K = 1728 (54 K-steps, 4 WMMAs each).
// A gathered once per K-step from {Xcat(past, stride 128), out0, out1} with
// zero padding (divergence only around loads; EXEC all-1s at every WMMA).
// Writes final NCHW f32 directly to d_out.
// ---------------------------------------------------------------------------
__global__ void k_gemm_fuse_wide(const unsigned short* __restrict__ Xcat,
                                 const unsigned short* __restrict__ out0,
                                 const unsigned short* __restrict__ out1,
                                 const unsigned short* __restrict__ Wf,
                                 const float* __restrict__ bias,
                                 float* __restrict__ out) {
  int wave = threadIdx.x >> 5;
  int ti   = (blockIdx.x * 4 + wave) << 4;
  int lane = threadIdx.x & 31;
  int half = lane >> 4, lan16 = lane & 15;
  // this lane's A row (pixel)
  long n = ti + lan16;
  int w = (int)(n % WW); long tt = n / WW;
  int h = (int)(tt % HH); int b = (int)(tt / HH);
  v8f acc[4] = {};
  for (int k0 = 0; k0 < KFUSE; k0 += 32) {
    int tap = k0 / 192, rem = k0 % 192;    // each 32-block lies in one tap/src
    int ky = tap / 3 - 1, kx = tap % 3 - 1;
    int src = rem >> 6, c0 = rem & 63;     // c0 in {0,32}
    int hh = h + ky, ww = w + kx;
    union { unsigned short u[16]; v16bf v; } am, bm[4];
    if ((unsigned)hh < (unsigned)HH && (unsigned)ww < (unsigned)WW) {
      long n2 = ((long)b * HH + hh) * WW + ww;
      const unsigned short* base =
          (src == 0) ? (Xcat + n2 * KRED + c0)
        : (src == 1) ? (out0 + n2 * CC + c0)
                     : (out1 + n2 * CC + c0);
#pragma unroll
      for (int j = 0; j < 16; ++j) {
        int tk = half * 16 + j;
        am.u[j] = base[tk];
      }
    } else {
#pragma unroll
      for (int j = 0; j < 16; ++j) am.u[j] = 0;   // zero padding row
    }
    const unsigned short* bbase = Wf + (long)lan16 * KFUSE + k0 + half * 16;
#pragma unroll
    for (int t = 0; t < 4; ++t) {
#pragma unroll
      for (int j = 0; j < 16; ++j) bm[t].u[j] = bbase[(long)t * 16 * KFUSE + j];
    }
#pragma unroll
    for (int t = 0; t < 4; ++t)
      acc[t] = __builtin_amdgcn_wmma_f32_16x16x32_bf16(false, am.v, false, bm[t].v,
                                                       (short)0, acc[t], false, false);
  }
#pragma unroll
  for (int t = 0; t < 4; ++t) {
    int j = t * 16 + lan16;                // output channel
    float bv = bias[j];
#pragma unroll
    for (int r = 0; r < 8; ++r) {
      long i = ti + r + 8 * half;          // output pixel
      int wo = (int)(i % WW); long ti2 = i / WW;
      int ho = (int)(ti2 % HH); int bo = (int)(ti2 / HH);
      out[(((long)bo * CC + j) * HH + ho) * WW + wo] = acc[t][r] + bv;
    }
  }
}

// ---------------------------------------------------------------------------
// Launch
// ---------------------------------------------------------------------------
extern "C" void kernel_launch(void* const* d_in, const int* in_sizes, int n_in,
                              void* d_out, int out_size, void* d_ws, size_t ws_size,
                              hipStream_t stream) {
  const float* past    = (const float*)d_in[0];
  const float* curr    = (const float*)d_in[1];
  const float* w_smooth= (const float*)d_in[2];
  const float* b_smooth= (const float*)d_in[3];
  const float* w_reduce= (const float*)d_in[4];
  const float* b_reduce= (const float*)d_in[5];
  const float* w_v0    = (const float*)d_in[6];
  const float* b_v0    = (const float*)d_in[7];
  const float* w_v1    = (const float*)d_in[8];
  const float* b_v1    = (const float*)d_in[9];
  const float* rpb0    = (const float*)d_in[10];
  const float* rpb1    = (const float*)d_in[11];
  const float* w_fuse  = (const float*)d_in[12];
  const float* b_fuse  = (const float*)d_in[13];
  float* outp = (float*)d_out;

  // workspace carve-up (256B aligned), total ~26.2 MB
  size_t off = 0;
  auto alloc = [&](size_t bytes) {
    void* p = (char*)d_ws + off;
    off += (bytes + 255) & ~(size_t)255;
    return p;
  };
  unsigned short* Xcat   = (unsigned short*)alloc((size_t)NPIX * KRED * 2);
  unsigned short* Wr     = (unsigned short*)alloc(64 * 128 * 2);
  unsigned short* Wv0    = (unsigned short*)alloc(64 * 64 * 2);
  unsigned short* Wv1    = (unsigned short*)alloc(64 * 64 * 2);
  unsigned short* Wf     = (unsigned short*)alloc(64 * KFUSE * 2);
  float*          past2f = (float*)alloc((size_t)NPIX * CC * 4);
  unsigned short* past2b = (unsigned short*)alloc((size_t)NPIX * CC * 2);
  float*          V0f    = (float*)alloc((size_t)NPIX * CC * 4);
  float*          V1f    = (float*)alloc((size_t)NPIX * CC * 4);
  unsigned short* out0b  = (unsigned short*)alloc((size_t)NPIX * CC * 2);
  unsigned short* out1b  = (unsigned short*)alloc((size_t)NPIX * CC * 2);
  (void)ws_size; (void)in_sizes; (void)n_in; (void)out_size;

  // 1) shift + grouped smooth conv + bf16 concat pack
  {
    int tot = BB * CC * HH * WW;
    k_shift_smooth_pack<<<(tot + 255) / 256, 256, 0, stream>>>(past, w_smooth, b_smooth, Xcat);
  }
  // 2) weight packing
  k_f32_to_bf16<<<(64 * 128 + 255) / 256, 256, 0, stream>>>(w_reduce, Wr, 64 * 128);
  k_f32_to_bf16<<<(64 * 64 + 255) / 256, 256, 0, stream>>>(w_v0, Wv0, 64 * 64);
  k_f32_to_bf16<<<(64 * 64 + 255) / 256, 256, 0, stream>>>(w_v1, Wv1, 64 * 64);
  k_pack_wfuse<<<(64 * KFUSE + 255) / 256, 256, 0, stream>>>(w_fuse, Wf);

  const int strips = NPIX / 16 / 4;            // 288 blocks x 4 waves
  // 3) reduce GEMM: past2 = [past||smooth] x w_reduce^T + b  (f32 + bf16 copies)
  k_gemm_bf16_wide<<<strips, 128, 0, stream>>>(Xcat, KRED, Wr, b_reduce, past2f, past2b, KRED);
  // 4) V projections
  k_gemm_bf16_wide<<<strips, 128, 0, stream>>>(past2b, CC, Wv0, b_v0, V0f, nullptr, CC);
  k_gemm_bf16_wide<<<strips, 128, 0, stream>>>(past2b, CC, Wv1, b_v1, V1f, nullptr, CC);
  // 5) neighborhood attention (ks = 3 and 7)
  k_natten<3><<<NPIX, 64, 0, stream>>>(curr, past2f, V0f, rpb0, out0b);
  k_natten<7><<<NPIX, 64, 0, stream>>>(curr, past2f, V1f, rpb1, out1b);
  // 6) fuse 3x3 conv as implicit GEMM (wide) -> final NCHW output
  k_gemm_fuse_wide<<<strips, 128, 0, stream>>>(Xcat, out0b, out1b, Wf, b_fuse, outp);
}
